// Module1_13125420057347
// MI455X (gfx1250) — matrix operand order repaired and checked by reading, not compiled, not run on hardware
//
#include <hip/hip_runtime.h>
#include <math.h>

// ---------------------------------------------------------------- constants
#define BATCH 2
#define CCH   128
#define HH    128
#define WW    128
#define HWSZ  16384
#define NHEAD 8
#define CHD   16
static const long long TSZ = (long long)BATCH * CCH * HWSZ; // 4,194,304 elems (16MB)

typedef __attribute__((ext_vector_type(16))) _Float16 v16h;
typedef __attribute__((ext_vector_type(4)))  _Float16 v4h;
typedef __attribute__((ext_vector_type(8)))  float    v8f;

static inline unsigned int gb(long long n, int b) { return (unsigned int)((n + (long long)b - 1) / b); }

// ================================================================ WMMA GEMM (fast)
// Guard-free batched strided GEMM: C[z] = alpha*A[z]xB[z] + beta*C[z] (+bias).
// Host guarantees: acs==1, ars%4==0, M%16==0, N%16==0, K%4==0, and
//   BK=true : brs==1, bcs%4==0 (B K-contiguous)
//   BK=false: bcs==1, brs%4==0 (B N-contiguous)
// One wave = one 16x16 tile; 64-wide K macro-tile -> 2x v_wmma_f32_16x16x32_f16.
// LDS tiles are fragment-ordered ([lane][h], k = 16*(lane/16) + h) so each lane reads its
// v16h fragment as one contiguous 32B LDS access (ds_load_b128 pair).
template <bool BK>
__global__ void k_bgemm_fast(const float* __restrict__ A, long long ars, long long abz,
                             const float* __restrict__ Bm, long long brs, long long bcs, long long bbz,
                             float* __restrict__ Cm, long long crs, long long ccs, long long cbz,
                             const float* __restrict__ bias, int biasMode,
                             float alpha, float beta, int N, int K)
{
    __shared__ __attribute__((aligned(32))) _Float16 sA[2][512];
    __shared__ __attribute__((aligned(32))) _Float16 sB[2][4][512];
    const int tid = threadIdx.x;
    const int wave = tid >> 5, lane = tid & 31;
    const int laneHi = lane >> 4, r16 = lane & 15;
    const int z = blockIdx.z;
    const float* Ab = A + (long long)z * abz;
    const float* Bb = Bm + (long long)z * bbz;
    float* Cb = Cm + (long long)z * cbz;
    const int m0 = blockIdx.y * 16;
    const int n0 = blockIdx.x * 64 + wave * 16;
    const bool active = (n0 < N);

    v8f acc = {};
    for (int k0 = 0; k0 < K; k0 += 64) {
        // ---- A tile 16x64: 256 quads over 128 threads ----
        for (int t = 0; t < 2; ++t) {
            int q = tid + 128 * t;
            int mm = q >> 4, kq = (q & 15) * 4;
            int u = kq >> 5, kk = kq & 31;
            int hi16 = (kk >> 4) << 4;
            int h0 = kk & 15;
            v4h hv = {};
            if (k0 + kq < K) {
                float4 f = *(const float4*)(Ab + (long long)(m0 + mm) * ars + (k0 + kq));
                hv[0] = (_Float16)f.x; hv[1] = (_Float16)f.y;
                hv[2] = (_Float16)f.z; hv[3] = (_Float16)f.w;
            }
            *(v4h*)&sA[u][(hi16 + mm) * 16 + h0] = hv;
        }
        // ---- B tile 64x16 per wave: 256 quads over 32 lanes ----
        if (active) {
            if (BK) {       // K contiguous: one ds_store_b64 per quad
                for (int t = 0; t < 8; ++t) {
                    int q = lane + 32 * t;
                    int nn = q >> 4, kq = (q & 15) * 4;
                    int u = kq >> 5, kk = kq & 31;
                    int hi16 = (kk >> 4) << 4;
                    int h0 = kk & 15;
                    v4h hv = {};
                    if (k0 + kq < K) {
                        float4 f = *(const float4*)(Bb + (long long)(n0 + nn) * bcs + (k0 + kq));
                        hv[0] = (_Float16)f.x; hv[1] = (_Float16)f.y;
                        hv[2] = (_Float16)f.z; hv[3] = (_Float16)f.w;
                    }
                    *(v4h*)&sB[u][wave][(hi16 + nn) * 16 + h0] = hv;
                }
            } else {        // N contiguous: float4 load, 4 b16 stores
                for (int t = 0; t < 8; ++t) {
                    int q = lane + 32 * t;
                    int kk6 = q >> 2, nn0 = (q & 3) * 4;
                    int u = kk6 >> 5, kk = kk6 & 31;
                    int hi16 = (kk >> 4) << 4;
                    int h = kk & 15;
                    float4 f = {0.f, 0.f, 0.f, 0.f};
                    if (k0 + kk6 < K)
                        f = *(const float4*)(Bb + (long long)(k0 + kk6) * brs + (n0 + nn0));
                    sB[u][wave][(hi16 + nn0 + 0) * 16 + h] = (_Float16)f.x;
                    sB[u][wave][(hi16 + nn0 + 1) * 16 + h] = (_Float16)f.y;
                    sB[u][wave][(hi16 + nn0 + 2) * 16 + h] = (_Float16)f.z;
                    sB[u][wave][(hi16 + nn0 + 3) * 16 + h] = (_Float16)f.w;
                }
            }
        }
        __syncthreads();
        if (k0 + 64 < K)  // hint next A tile into cache (global_prefetch_b8)
            __builtin_prefetch(&Ab[(long long)m0 * ars + (k0 + 64)], 0, 1);

        v16h a0 = *(const v16h*)&sA[0][lane * 16];
        v16h b0 = *(const v16h*)&sB[0][wave][lane * 16];
        acc = __builtin_amdgcn_wmma_f32_16x16x32_f16(false, a0, false, b0,
                                                     (short)0, acc, false, false);
        v16h a1 = *(const v16h*)&sA[1][lane * 16];
        v16h b1 = *(const v16h*)&sB[1][wave][lane * 16];
        acc = __builtin_amdgcn_wmma_f32_16x16x32_f16(false, a1, false, b1,
                                                     (short)0, acc, false, false);
        __syncthreads();
    }
    if (active) {
        __attribute__((aligned(32))) float accA[8];
        *(v8f*)accA = acc;
        int gn = n0 + r16;
        for (int v = 0; v < 8; ++v) {
            int gm = m0 + v + laneHi * 8;
            long long off = (long long)gm * crs + (long long)gn * ccs;
            float val = alpha * accA[v];
            if (beta != 0.f) val += beta * Cb[off];
            if (biasMode == 1) val += bias[gn];
            else if (biasMode == 2) val += bias[gm];
            Cb[off] = val;
        }
    }
}

// ================================================================ WMMA GEMM (generic)
// Guarded scalar loaders; used only for tiny/odd tiles (e.g. SE-gate conv1, M=8).
__global__ void k_bgemm_gen(const float* __restrict__ A, long long ars, long long acs, long long abz,
                            const float* __restrict__ Bm, long long brs, long long bcs, long long bbz,
                            float* __restrict__ Cm, long long crs, long long ccs, long long cbz,
                            const float* __restrict__ bias, int biasMode,
                            float alpha, float beta, int M, int N, int K)
{
    __shared__ __attribute__((aligned(32))) _Float16 sA[2][512];
    __shared__ __attribute__((aligned(32))) _Float16 sB[2][4][512];
    const int tid = threadIdx.x;
    const int wave = tid >> 5, lane = tid & 31;
    const int laneHi = lane >> 4, r16 = lane & 15;
    const int z = blockIdx.z;
    const float* Ab = A + (long long)z * abz;
    const float* Bb = Bm + (long long)z * bbz;
    float* Cb = Cm + (long long)z * cbz;
    const int m0 = blockIdx.y * 16;
    const int n0 = blockIdx.x * 64 + wave * 16;
    const bool active = (n0 < N);

    v8f acc = {};
    for (int k0 = 0; k0 < K; k0 += 64) {
        for (int t = 0; t < 8; ++t) {
            int e = tid + 128 * t;              // 0..1023
            int mm = e >> 6, kk6 = e & 63;
            int u = kk6 >> 5, kk = kk6 & 31;
            float v = 0.f;
            int gm = m0 + mm, gk = k0 + kk6;
            if (gm < M && gk < K) v = Ab[(long long)gm * ars + (long long)gk * acs];
            int hi16 = (kk >> 4) << 4;
            int h0 = kk & 15;
            sA[u][(hi16 + mm) * 16 + h0] = (_Float16)v;
        }
        if (active) {
            for (int t = 0; t < 32; ++t) {
                int e = lane + 32 * t;          // 0..1023
                int kk6 = e >> 4, nn = e & 15;
                int u = kk6 >> 5, kk = kk6 & 31;
                float v = 0.f;
                int gk = k0 + kk6, gn = n0 + nn;
                if (gk < K && gn < N) v = Bb[(long long)gk * brs + (long long)gn * bcs];
                int hi16 = (kk >> 4) << 4;
                sB[u][wave][(hi16 + nn) * 16 + (kk & 15)] = (_Float16)v;
            }
        }
        __syncthreads();
        v16h a0 = *(const v16h*)&sA[0][lane * 16];
        v16h b0 = *(const v16h*)&sB[0][wave][lane * 16];
        acc = __builtin_amdgcn_wmma_f32_16x16x32_f16(false, a0, false, b0,
                                                     (short)0, acc, false, false);
        v16h a1 = *(const v16h*)&sA[1][lane * 16];
        v16h b1 = *(const v16h*)&sB[1][wave][lane * 16];
        acc = __builtin_amdgcn_wmma_f32_16x16x32_f16(false, a1, false, b1,
                                                     (short)0, acc, false, false);
        __syncthreads();
    }
    __attribute__((aligned(32))) float accA[8];
    *(v8f*)accA = acc;
    for (int v = 0; v < 8; ++v) {
        int gm = m0 + v + laneHi * 8;
        int gn = n0 + r16;
        if (gm < M && gn < N) {
            long long off = (long long)gm * crs + (long long)gn * ccs;
            float val = alpha * accA[v];
            if (beta != 0.f) val += beta * Cb[off];
            if (biasMode == 1) val += bias[gn];
            else if (biasMode == 2) val += bias[gm];
            Cb[off] = val;
        }
    }
}

// ================================================================ misc kernels
__global__ void k_layernorm(const float* __restrict__ x, const float* __restrict__ w,
                            const float* __restrict__ b, float* __restrict__ y)
{
    long long p = (long long)blockIdx.x * blockDim.x + threadIdx.x;
    if (p >= (long long)BATCH * HWSZ) return;
    long long bb = p / HWSZ, hw = p % HWSZ;
    long long base = bb * CCH * HWSZ + hw;
    float s = 0.f, s2 = 0.f;
    for (int c = 0; c < CCH; ++c) {
        float v = x[base + (long long)c * HWSZ];
        s += v; s2 += v * v;
    }
    float mu = s / CCH;
    float var = s2 / CCH - mu * mu;
    float rs = rsqrtf(var + 1e-5f);
    for (int c = 0; c < CCH; ++c) {
        float v = x[base + (long long)c * HWSZ];
        y[base + (long long)c * HWSZ] = (v - mu) * rs * w[c] + b[c];
    }
}

__global__ void k_init_tables(float* Fr, float* Fi, float* iFr, float* iFi,
                              float* iF16r, float* iF16i, float* twr, float* twi)
{
    int i = blockIdx.x * blockDim.x + threadIdx.x;
    if (i < 16384) {
        int k = i >> 7, n = i & 127;
        int kn = (k * n) & 127;
        double th = 6.283185307179586 * (double)kn / 128.0;
        Fr[i]  = (float)cos(th);  Fi[i]  = (float)(-sin(th));
        iFr[i] = (float)(cos(th) / 128.0);
        iFi[i] = (float)(sin(th) / 128.0);
        int cc = i >> 7, bb = i & 127;
        int m = (cc * bb) & 16383;
        double ph = 6.283185307179586 * (double)m / 16384.0;
        twr[i] = (float)cos(ph); twi[i] = (float)sin(ph);
    }
    if (i < 256) {
        int k = i >> 4, n = i & 15;
        int m = (k * n) & 15;
        double th = 6.283185307179586 * (double)m / 16.0;
        iF16r[i] = (float)(cos(th) / 16.0);
        iF16i[i] = (float)(sin(th) / 16.0);
    }
}

__global__ void k_bn_stats(const float* __restrict__ x, int Cm, float* mu, float* rstd)
{
    __shared__ float r1[256], r2[256];
    int c = blockIdx.x;
    float s = 0.f, s2 = 0.f;
    const int N = BATCH * HWSZ;
    for (int i = threadIdx.x; i < N; i += 256) {
        int b = i / HWSZ, hw = i % HWSZ;
        float v = x[((long long)b * Cm + c) * HWSZ + hw];
        s += v; s2 += v * v;
    }
    r1[threadIdx.x] = s; r2[threadIdx.x] = s2; __syncthreads();
    for (int w = 128; w > 0; w >>= 1) {
        if (threadIdx.x < w) { r1[threadIdx.x] += r1[threadIdx.x + w]; r2[threadIdx.x] += r2[threadIdx.x + w]; }
        __syncthreads();
    }
    if (threadIdx.x == 0) {
        float m = r1[0] / N, var = r2[0] / N - m * m;
        mu[c] = m; rstd[c] = rsqrtf(var + 1e-5f);
    }
}

__global__ void k_bn_relu(const float* __restrict__ x, float* __restrict__ y,
                          const float* mu, const float* rstd,
                          const float* g, const float* b, int Cm, long long n)
{
    long long i = (long long)blockIdx.x * blockDim.x + threadIdx.x;
    if (i >= n) return;
    int c = (int)((i / HWSZ) % Cm);
    float v = (x[i] - mu[c]) * rstd[c] * g[c] + b[c];
    y[i] = fmaxf(v, 0.f);
}

__global__ void k_sig_cmul(const float* __restrict__ g, const float* __restrict__ fr,
                           const float* __restrict__ fi, float* __restrict__ outr,
                           float* __restrict__ outi, long long n)
{
    long long i = (long long)blockIdx.x * blockDim.x + threadIdx.x;
    if (i >= n) return;
    float s = 1.f / (1.f + expf(-g[i]));
    outr[i] = s * fr[i];
    outi[i] = s * fi[i];
}

__device__ __forceinline__ float gglu_f(float x)
{
    float ge = 0.5f * x * (1.f + erff(x * 0.70710678118654752f));
    return ge * x;
}

__global__ void k_gate_abs_gglu_embed(const float* __restrict__ g, const float* __restrict__ fr,
                                      const float* __restrict__ fi, float* __restrict__ out,
                                      int Cs, int Ct, int coff, long long n)
{
    long long i = (long long)blockIdx.x * blockDim.x + threadIdx.x;
    if (i >= n) return;
    long long b = i / ((long long)Cs * HWSZ);
    long long r = i % ((long long)Cs * HWSZ);
    long long c = r / HWSZ, hw = r % HWSZ;
    float s = 1.f / (1.f + expf(-g[i]));
    float m = s * sqrtf(fr[i] * fr[i] + fi[i] * fi[i]);
    out[((b * Ct) + coff + c) * HWSZ + hw] = gglu_f(m);
}

__global__ void k_gglu_embed(float* __restrict__ buf, int Cs, int Ct, int coff, long long n)
{
    long long i = (long long)blockIdx.x * blockDim.x + threadIdx.x;
    if (i >= n) return;
    long long b = i / ((long long)Cs * HWSZ);
    long long r = i % ((long long)Cs * HWSZ);
    long long c = r / HWSZ, hw = r % HWSZ;
    long long pos = ((b * Ct) + coff + c) * HWSZ + hw;
    buf[pos] = gglu_f(buf[pos]);
}

__global__ void k_abs_embed(const float* __restrict__ xr, const float* __restrict__ xi,
                            float* __restrict__ out, int Cs, int Ct, int coff, long long n)
{
    long long i = (long long)blockIdx.x * blockDim.x + threadIdx.x;
    if (i >= n) return;
    long long b = i / ((long long)Cs * HWSZ);
    long long r = i % ((long long)Cs * HWSZ);
    long long c = r / HWSZ, hw = r % HWSZ;
    out[((b * Ct) + coff + c) * HWSZ + hw] = sqrtf(xr[i] * xr[i] + xi[i] * xi[i]);
}

__global__ void k_copy_embed(const float* __restrict__ in, float* __restrict__ out,
                             int Cs, int Ct, int coff, long long n)
{
    long long i = (long long)blockIdx.x * blockDim.x + threadIdx.x;
    if (i >= n) return;
    long long b = i / ((long long)Cs * HWSZ);
    long long r = i % ((long long)Cs * HWSZ);
    long long c = r / HWSZ, hw = r % HWSZ;
    out[((b * Ct) + coff + c) * HWSZ + hw] = in[i];
}

__global__ void k_l2norm_rows(const float* __restrict__ inr, const float* __restrict__ ini,
                              float* __restrict__ outr, float* __restrict__ outi, int n)
{
    __shared__ float red[256];
    long long z = blockIdx.x;
    const float* ir = inr + z * n;
    const float* ii = ini ? ini + z * n : nullptr;
    float s = 0.f;
    for (int i = threadIdx.x; i < n; i += 256) {
        float a = ir[i]; s += a * a;
        if (ii) { float b = ii[i]; s += b * b; }
    }
    red[threadIdx.x] = s; __syncthreads();
    for (int w = 128; w > 0; w >>= 1) {
        if (threadIdx.x < w) red[threadIdx.x] += red[threadIdx.x + w];
        __syncthreads();
    }
    float inv = 1.f / fmaxf(sqrtf(red[0]), 1e-12f);
    for (int i = threadIdx.x; i < n; i += 256) {
        outr[z * n + i] = ir[i] * inv;
        if (ii) outi[z * n + i] = ii[i] * inv;
    }
}

// softmax over last dim of (bh,16,16) matrices; scales logits by temp[h] first.
__global__ void k_softmax16(float* mr, float* mi, const float* temp, int nbh)
{
    int z = blockIdx.x; if (z >= nbh) return;
    float t = temp[z % NHEAD];
    int r = threadIdx.x;
    if (r < 16) {
        for (int pass = 0; pass < 2; ++pass) {
            float* m = pass ? mi : mr;
            if (!m) continue;
            long long base = (long long)z * 256 + r * 16;
            float v[16]; float mx = -1e30f;
            for (int d = 0; d < 16; ++d) { v[d] = m[base + d] * t; mx = fmaxf(mx, v[d]); }
            float s = 0.f;
            for (int d = 0; d < 16; ++d) { v[d] = expf(v[d] - mx); s += v[d]; }
            float inv = 1.f / s;
            for (int d = 0; d < 16; ++d) m[base + d] = v[d] * inv;
        }
    }
}

__global__ void k_ctwiddle(float* __restrict__ tr, float* __restrict__ ti,
                           const float* __restrict__ twr, const float* __restrict__ twi, long long n)
{
    long long i = (long long)blockIdx.x * blockDim.x + threadIdx.x;
    if (i >= n) return;
    int e = (int)(i & 16383);
    float a = tr[i], b = ti[i], c = twr[e], d = twi[e];
    tr[i] = a * c - b * d;
    ti[i] = a * d + b * c;
}

__global__ void k_transpose128c(const float* __restrict__ ir, const float* __restrict__ ii,
                                float* __restrict__ outr, float* __restrict__ outi, long long n)
{
    long long i = (long long)blockIdx.x * blockDim.x + threadIdx.x;
    if (i >= n) return;
    long long z = i >> 14;
    int e = (int)(i & 16383);
    int r = e >> 7, c = e & 127;
    long long src = (z << 14) + c * 128 + r;
    outr[i] = ir[src];
    outi[i] = ii[src];
}

__global__ void k_grouped_conv(const float* __restrict__ in, int Cin,
                               const float* __restrict__ wgt,
                               int Cout, int groups, int ks, int pad,
                               float* __restrict__ out, int Ctout, int coff)
{
    long long idx = (long long)blockIdx.x * blockDim.x + threadIdx.x;
    long long total = (long long)BATCH * Cout * HWSZ;
    if (idx >= total) return;
    int hw = (int)(idx % HWSZ);
    long long t = idx / HWSZ;
    int oc = (int)(t % Cout);
    int b  = (int)(t / Cout);
    int h = hw / WW, w = hw % WW;
    int inPer = Cin / groups, outPer = Cout / groups;
    int gi = oc / outPer;
    float acc = 0.f;
    for (int j = 0; j < inPer; ++j) {
        int ic = gi * inPer + j;
        const float* ip = in + ((long long)b * Cin + ic) * HWSZ;
        const float* wp = wgt + (long long)(oc * inPer + j) * ks * ks;
        for (int ky = 0; ky < ks; ++ky) {
            int iy = h + ky - pad; if (iy < 0 || iy >= HH) continue;
            for (int kx = 0; kx < ks; ++kx) {
                int ix = w + kx - pad; if (ix < 0 || ix >= WW) continue;
                acc += ip[iy * WW + ix] * wp[ky * ks + kx];
            }
        }
    }
    out[((long long)b * Ctout + coff + oc) * HWSZ + hw] = acc;
}

__global__ void k_add3(const float* a, const float* b, const float* c, float* o, long long n)
{
    long long i = (long long)blockIdx.x * blockDim.x + threadIdx.x;
    if (i < n) o[i] = a[i] + b[i] + c[i];
}
__global__ void k_add2(const float* a, const float* b, float* o, long long n)
{
    long long i = (long long)blockIdx.x * blockDim.x + threadIdx.x;
    if (i < n) o[i] = a[i] + b[i];
}

// ================================================================ host helpers
static void bgemm(hipStream_t s,
                  const float* A, long long ars, long long acs, long long abz,
                  const float* B, long long brs, long long bcs, long long bbz,
                  float* C, long long crs, long long ccs, long long cbz,
                  int batch, int M, int N, int K,
                  float alpha, float beta, const float* bias, int biasMode)
{
    dim3 g((N + 63) / 64, (M + 15) / 16, batch);
    bool fastOK = (acs == 1) && !(ars & 3LL) && !(M & 15) && !(N & 15) && !(K & 3);
    if (fastOK && brs == 1 && !(bcs & 3LL)) {
        k_bgemm_fast<true><<<g, 128, 0, s>>>(A, ars, abz, B, brs, bcs, bbz,
                                             C, crs, ccs, cbz, bias, biasMode, alpha, beta, N, K);
    } else if (fastOK && bcs == 1 && !(brs & 3LL)) {
        k_bgemm_fast<false><<<g, 128, 0, s>>>(A, ars, abz, B, brs, bcs, bbz,
                                              C, crs, ccs, cbz, bias, biasMode, alpha, beta, N, K);
    } else {
        k_bgemm_gen<<<g, 128, 0, s>>>(A, ars, acs, abz, B, brs, bcs, bbz,
                                      C, crs, ccs, cbz, bias, biasMode, alpha, beta, M, N, K);
    }
}

// 1x1 conv in NCHW as batched GEMM: out[b,oc+coff,hw] = sum_ic w[oc,ic] in[b,ic,hw] + bias
static void conv1x1(hipStream_t s, const float* in, int Cin,
                    const float* w, const float* bias,
                    float* out, int Cout, int Ctot, int coff)
{
    bgemm(s, w, Cin, 1, 0,
          in, HWSZ, 1, (long long)Cin * HWSZ,
          out + (long long)coff * HWSZ, HWSZ, 1, (long long)Ctot * HWSZ,
          BATCH, Cout, HWSZ, Cin, 1.f, 0.f, bias, bias ? 2 : 0);
}

// 2D DFT over (h,w) of real NCHW input: row DFT (Y = X*F) then col DFT (Z = F*Y)
static void fft2_real(hipStream_t s, const float* x, float* rowR, float* rowI,
                      float* outR, float* outI, int Cn, const float* Fr, const float* Fi)
{
    int rows = BATCH * Cn * HH;
    bgemm(s, x, WW, 1, 0, Fr, WW, 1, 0, rowR, WW, 1, 0, 1, rows, WW, WW, 1.f, 0.f, 0, 0);
    bgemm(s, x, WW, 1, 0, Fi, WW, 1, 0, rowI, WW, 1, 0, 1, rows, WW, WW, 1.f, 0.f, 0, 0);
    int bt = BATCH * Cn;
    bgemm(s, Fr, WW, 1, 0, rowR, WW, 1, HWSZ, outR, WW, 1, HWSZ, bt, HH, WW, HH, 1.f, 0.f, 0, 0);
    bgemm(s, Fi, WW, 1, 0, rowI, WW, 1, HWSZ, outR, WW, 1, HWSZ, bt, HH, WW, HH, -1.f, 1.f, 0, 0);
    bgemm(s, Fr, WW, 1, 0, rowI, WW, 1, HWSZ, outI, WW, 1, HWSZ, bt, HH, WW, HH, 1.f, 0.f, 0, 0);
    bgemm(s, Fi, WW, 1, 0, rowR, WW, 1, HWSZ, outI, WW, 1, HWSZ, bt, HH, WW, HH, 1.f, 1.f, 0, 0);
}

// 2D DFT over (h,w) of complex input (used with iF matrices for ifft2).
// out may alias in (in is dead after row stage).
static void fft2_cplx(hipStream_t s, const float* xr, const float* xi,
                      float* rowR, float* rowI, float* outR, float* outI,
                      int Cn, const float* Fr, const float* Fi)
{
    int rows = BATCH * Cn * HH;
    bgemm(s, xr, WW, 1, 0, Fr, WW, 1, 0, rowR, WW, 1, 0, 1, rows, WW, WW, 1.f, 0.f, 0, 0);
    bgemm(s, xi, WW, 1, 0, Fi, WW, 1, 0, rowR, WW, 1, 0, 1, rows, WW, WW, -1.f, 1.f, 0, 0);
    bgemm(s, xr, WW, 1, 0, Fi, WW, 1, 0, rowI, WW, 1, 0, 1, rows, WW, WW, 1.f, 0.f, 0, 0);
    bgemm(s, xi, WW, 1, 0, Fr, WW, 1, 0, rowI, WW, 1, 0, 1, rows, WW, WW, 1.f, 1.f, 0, 0);
    int bt = BATCH * Cn;
    bgemm(s, Fr, WW, 1, 0, rowR, WW, 1, HWSZ, outR, WW, 1, HWSZ, bt, HH, WW, HH, 1.f, 0.f, 0, 0);
    bgemm(s, Fi, WW, 1, 0, rowI, WW, 1, HWSZ, outR, WW, 1, HWSZ, bt, HH, WW, HH, -1.f, 1.f, 0, 0);
    bgemm(s, Fr, WW, 1, 0, rowI, WW, 1, HWSZ, outI, WW, 1, HWSZ, bt, HH, WW, HH, 1.f, 0.f, 0, 0);
    bgemm(s, Fi, WW, 1, 0, rowR, WW, 1, HWSZ, outI, WW, 1, HWSZ, bt, HH, WW, HH, 1.f, 1.f, 0, 0);
}

// ifft2 over (c=16, n=16384) of a (B,NHEAD,16,16384) complex tensor (in place, t1 scratch).
// 16384-point inverse DFT is Cooley-Tukey 128x128: GEMM, twiddle, GEMM, transpose; then 16-pt GEMM.
static void ifft_cn(hipStream_t s, float* inR, float* inI, float* t1R, float* t1I,
                    const float* iFr, const float* iFi,
                    const float* iF16r, const float* iF16i,
                    const float* twr, const float* twi)
{
    const int bt = BATCH * NHEAD * CHD;                // 256 matrices of 128x128
    bgemm(s, iFr, 128, 1, 0, inR, 128, 1, 16384, t1R, 128, 1, 16384, bt, 128, 128, 128, 1.f, 0.f, 0, 0);
    bgemm(s, iFi, 128, 1, 0, inI, 128, 1, 16384, t1R, 128, 1, 16384, bt, 128, 128, 128, -1.f, 1.f, 0, 0);
    bgemm(s, iFr, 128, 1, 0, inI, 128, 1, 16384, t1I, 128, 1, 16384, bt, 128, 128, 128, 1.f, 0.f, 0, 0);
    bgemm(s, iFi, 128, 1, 0, inR, 128, 1, 16384, t1I, 128, 1, 16384, bt, 128, 128, 128, 1.f, 1.f, 0, 0);
    long long n = (long long)bt * 16384;
    k_ctwiddle<<<gb(n, 256), 256, 0, s>>>(t1R, t1I, twr, twi, n);
    bgemm(s, t1R, 128, 1, 16384, iFr, 128, 1, 0, inR, 128, 1, 16384, bt, 128, 128, 128, 1.f, 0.f, 0, 0);
    bgemm(s, t1I, 128, 1, 16384, iFi, 128, 1, 0, inR, 128, 1, 16384, bt, 128, 128, 128, -1.f, 1.f, 0, 0);
    bgemm(s, t1R, 128, 1, 16384, iFi, 128, 1, 0, inI, 128, 1, 16384, bt, 128, 128, 128, 1.f, 0.f, 0, 0);
    bgemm(s, t1I, 128, 1, 16384, iFr, 128, 1, 0, inI, 128, 1, 16384, bt, 128, 128, 128, 1.f, 1.f, 0, 0);
    k_transpose128c<<<gb(n, 256), 256, 0, s>>>(inR, inI, t1R, t1I, n);
    const int bh = BATCH * NHEAD;
    const long long bs = (long long)CHD * 16384;
    bgemm(s, iF16r, 16, 1, 0, t1R, 16384, 1, bs, inR, 16384, 1, bs, bh, 16, 16384, 16, 1.f, 0.f, 0, 0);
    bgemm(s, iF16i, 16, 1, 0, t1I, 16384, 1, bs, inR, 16384, 1, bs, bh, 16, 16384, 16, -1.f, 1.f, 0, 0);
    bgemm(s, iF16r, 16, 1, 0, t1I, 16384, 1, bs, inI, 16384, 1, bs, bh, 16, 16384, 16, 1.f, 0.f, 0, 0);
    bgemm(s, iF16i, 16, 1, 0, t1R, 16384, 1, bs, inI, 16384, 1, bs, bh, 16, 16384, 16, 1.f, 1.f, 0, 0);
}

// SE gate (pre-sigmoid): conv1x1 -> BN(train) -> ReLU -> conv1x1 + bias
static void se_gate_h(hipStream_t s, const float* fin, int Cg,
                      const float* w1, const float* b1,
                      const float* bng, const float* bnb,
                      const float* w2, const float* b2, int Cout,
                      float* gout, float* glin, float* gmid, float* bmu, float* brs)
{
    conv1x1(s, fin, Cg, w1, b1, glin, 8, 8, 0);
    k_bn_stats<<<8, 256, 0, s>>>(glin, 8, bmu, brs);
    long long n = (long long)BATCH * 8 * HWSZ;
    k_bn_relu<<<gb(n, 256), 256, 0, s>>>(glin, gmid, bmu, brs, bng, bnb, 8, n);
    conv1x1(s, gmid, 8, w2, b2, gout, Cout, Cout, 0);
}

// ================================================================ inputs (insertion order)
enum {
    IN_X = 0, IN_N1W, IN_N1B, IN_N2W, IN_N2B,
    IN_AF_TEMP, IN_AF_PROJ,
    IN_AF_G_W1, IN_AF_G_B1, IN_AF_G_BNG, IN_AF_G_BNB, IN_AF_G_W2, IN_AF_G_B2,
    IN_AS_TEMP, IN_AS_Q1W, IN_AS_Q1B, IN_AS_K1W, IN_AS_K1B, IN_AS_V1W, IN_AS_V1B,
    IN_AS_Q3, IN_AS_Q5, IN_AS_K3, IN_AS_K5, IN_AS_V3, IN_AS_V5, IN_AS_C3, IN_AS_C5, IN_AS_PROJ,
    IN_FF_DW1, IN_FF_DW3, IN_FF_DW5, IN_FF_DW7, IN_FF_PROJ,
    IN_FF_G0_W1, IN_FF_G0_B1, IN_FF_G0_BNG, IN_FF_G0_BNB, IN_FF_G0_W2, IN_FF_G0_B2,
    IN_FF_G1_W1, IN_FF_G1_B1, IN_FF_G1_BNG, IN_FF_G1_BNB, IN_FF_G1_W2, IN_FF_G1_B2
};

extern "C" void kernel_launch(void* const* d_in, const int* in_sizes, int n_in,
                              void* d_out, int out_size, void* d_ws, size_t ws_size,
                              hipStream_t stream)
{
    (void)in_sizes; (void)n_in; (void)out_size; (void)ws_size;
    hipStream_t st = stream;
    const float* x = (const float*)d_in[IN_X];
    auto P = [&](int i) { return (const float*)d_in[i]; };

    float* ws = (float*)d_ws;
    auto S = [&](int i) { return ws + (long long)i * TSZ; };   // 16MB slots
    // small tables region after slot 21
    float* SM     = ws + 21 * TSZ;
    float* F128r  = SM;             float* F128i  = F128r + 16384;
    float* iF128r = F128i + 16384;  float* iF128i = iF128r + 16384;
    float* iF16r  = iF128i + 16384; float* iF16i  = iF16r + 256;
    float* twr    = iF16i + 256;    float* twi    = twr + 16384;
    float* attnR  = twi + 16384;    float* attnI  = attnR + 4096;
    float* glin   = attnI + 4096;   float* gmid   = glin + (long long)BATCH * 8 * HWSZ;
    float* bmu    = gmid + (long long)BATCH * 8 * HWSZ; float* brs = bmu + 8;

    const long long HS = (long long)CHD * HWSZ;       // 262144 per (b,h)
    const long long N1 = TSZ, N2 = 2 * TSZ;

    k_init_tables<<<gb(16384, 256), 256, 0, st>>>(F128r, F128i, iF128r, iF128i, iF16r, iF16i, twr, twi);

    // ---------------- y = LN1(x) -> s0
    k_layernorm<<<gb((long long)BATCH * HWSZ, 256), 256, 0, st>>>(x, P(IN_N1W), P(IN_N1B), S(0));

    // ================= attn_F =================
    // f = fft2(y): row stage s3/s4, f -> s1/s2
    fft2_real(st, S(0), S(3), S(4), S(1), S(2), CCH, F128r, F128i);
    // q = l2norm(f) per (b,h,c) row over n=16384 -> s5/s6
    k_l2norm_rows<<<BATCH * NHEAD * CHD, 256, 0, st>>>(S(1), S(2), S(5), S(6), HWSZ);
    // attn = q . q^T (complex, no conj), per (b,h)
    bgemm(st, S(5), HWSZ, 1, HS, S(5), 1, HWSZ, HS, attnR, 16, 1, 256, 16, 16, 16, HWSZ, 1.f, 0.f, 0, 0);
    bgemm(st, S(6), HWSZ, 1, HS, S(6), 1, HWSZ, HS, attnR, 16, 1, 256, 16, 16, 16, HWSZ, -1.f, 1.f, 0, 0);
    bgemm(st, S(5), HWSZ, 1, HS, S(6), 1, HWSZ, HS, attnI, 16, 1, 256, 16, 16, 16, HWSZ, 1.f, 0.f, 0, 0);
    bgemm(st, S(6), HWSZ, 1, HS, S(5), 1, HWSZ, HS, attnI, 16, 1, 256, 16, 16, 16, HWSZ, 1.f, 1.f, 0, 0);
    k_softmax16<<<BATCH * NHEAD, 32, 0, st>>>(attnR, attnI, P(IN_AF_TEMP), BATCH * NHEAD);
    // e = attn . v (v = f), complex -> s3/s4
    bgemm(st, attnR, 16, 1, 256, S(1), HWSZ, 1, HS, S(3), HWSZ, 1, HS, 16, 16, HWSZ, 16, 1.f, 0.f, 0, 0);
    bgemm(st, attnI, 16, 1, 256, S(2), HWSZ, 1, HS, S(3), HWSZ, 1, HS, 16, 16, HWSZ, 16, -1.f, 1.f, 0, 0);
    bgemm(st, attnR, 16, 1, 256, S(2), HWSZ, 1, HS, S(4), HWSZ, 1, HS, 16, 16, HWSZ, 16, 1.f, 0.f, 0, 0);
    bgemm(st, attnI, 16, 1, 256, S(1), HWSZ, 1, HS, S(4), HWSZ, 1, HS, 16, 16, HWSZ, 16, 1.f, 1.f, 0, 0);
    // ifft2 over (c, n), abs -> catF (slots 7-8, 256 channels) [0:128)
    ifft_cn(st, S(3), S(4), S(5), S(6), iF128r, iF128i, iF16r, iF16i, twr, twi);
    k_abs_embed<<<gb(N1, 256), 256, 0, st>>>(S(3), S(4), S(7), CCH, 2 * CCH, 0, N1);
    // out_l: se_gate(f.real) * f, ifft2 spatial, abs -> catF [128:256)
    se_gate_h(st, S(1), CCH, P(IN_AF_G_W1), P(IN_AF_G_B1), P(IN_AF_G_BNG), P(IN_AF_G_BNB),
              P(IN_AF_G_W2), P(IN_AF_G_B2), CCH, S(3), glin, gmid, bmu, brs);
    k_sig_cmul<<<gb(N1, 256), 256, 0, st>>>(S(3), S(1), S(2), S(5), S(6), N1);
    fft2_cplx(st, S(5), S(6), S(3), S(4), S(5), S(6), CCH, iF128r, iF128i);
    k_abs_embed<<<gb(N1, 256), 256, 0, st>>>(S(5), S(6), S(7), CCH, 2 * CCH, CCH, N1);
    // attnF_out = proj(catF) -> s9
    conv1x1(st, S(7), 2 * CCH, P(IN_AF_PROJ), nullptr, S(9), CCH, CCH, 0);

    // ================= attn_S ================= (y = s0)
    // q
    conv1x1(st, S(0), CCH, P(IN_AS_Q1W), P(IN_AS_Q1B), S(1), CCH, CCH, 0);
    k_grouped_conv<<<gb((long long)BATCH * 64 * HWSZ, 256), 256, 0, st>>>(S(1), CCH, P(IN_AS_Q3), 64, 64, 3, 1, S(2), CCH, 0);
    k_grouped_conv<<<gb((long long)BATCH * 64 * HWSZ, 256), 256, 0, st>>>(S(1), CCH, P(IN_AS_Q5), 64, 64, 5, 2, S(2), CCH, 64);
    // k
    conv1x1(st, S(0), CCH, P(IN_AS_K1W), P(IN_AS_K1B), S(1), CCH, CCH, 0);
    k_grouped_conv<<<gb((long long)BATCH * 64 * HWSZ, 256), 256, 0, st>>>(S(1), CCH, P(IN_AS_K3), 64, 64, 3, 1, S(3), CCH, 0);
    k_grouped_conv<<<gb((long long)BATCH * 64 * HWSZ, 256), 256, 0, st>>>(S(1), CCH, P(IN_AS_K5), 64, 64, 5, 2, S(3), CCH, 64);
    // v
    conv1x1(st, S(0), CCH, P(IN_AS_V1W), P(IN_AS_V1B), S(1), CCH, CCH, 0);
    k_grouped_conv<<<gb((long long)BATCH * 64 * HWSZ, 256), 256, 0, st>>>(S(1), CCH, P(IN_AS_V3), 64, 64, 3, 1, S(4), CCH, 0);
    k_grouped_conv<<<gb((long long)BATCH * 64 * HWSZ, 256), 256, 0, st>>>(S(1), CCH, P(IN_AS_V5), 64, 64, 5, 2, S(4), CCH, 64);
    // l2norm q,k (real, in place)
    k_l2norm_rows<<<BATCH * NHEAD * CHD, 256, 0, st>>>(S(2), nullptr, S(2), nullptr, HWSZ);
    k_l2norm_rows<<<BATCH * NHEAD * CHD, 256, 0, st>>>(S(3), nullptr, S(3), nullptr, HWSZ);
    // attn = softmax(q.k^T * temp); out = attn.v -> s5
    bgemm(st, S(2), HWSZ, 1, HS, S(3), 1, HWSZ, HS, attnR, 16, 1, 256, 16, 16, 16, HWSZ, 1.f, 0.f, 0, 0);
    k_softmax16<<<BATCH * NHEAD, 32, 0, st>>>(attnR, nullptr, P(IN_AS_TEMP), BATCH * NHEAD);
    bgemm(st, attnR, 16, 1, 256, S(4), HWSZ, 1, HS, S(5), HWSZ, 1, HS, 16, 16, HWSZ, 16, 1.f, 0.f, 0, 0);
    // catS (slots 1-2): out | out_l
    k_copy_embed<<<gb(N1, 256), 256, 0, st>>>(S(5), S(1), CCH, 2 * CCH, 0, N1);
    k_grouped_conv<<<gb((long long)BATCH * 64 * HWSZ, 256), 256, 0, st>>>(S(0), CCH, P(IN_AS_C3), 64, 64, 3, 1, S(1), 2 * CCH, CCH);
    k_grouped_conv<<<gb((long long)BATCH * 64 * HWSZ, 256), 256, 0, st>>>(S(0), CCH, P(IN_AS_C5), 64, 64, 5, 2, S(1), 2 * CCH, CCH + 64);
    conv1x1(st, S(1), 2 * CCH, P(IN_AS_PROJ), nullptr, S(3), CCH, CCH, 0);
    // x1 = x + attnF_out + attnS_out -> s4
    k_add3<<<gb(N1, 256), 256, 0, st>>>(x, S(9), S(3), S(4), N1);

    // ================= FFN =================
    k_layernorm<<<gb((long long)BATCH * HWSZ, 256), 256, 0, st>>>(S(4), P(IN_N2W), P(IN_N2B), S(0));
    // f2 = fft2(y2) -> s5/s6
    fft2_real(st, S(0), S(1), S(2), S(5), S(6), CCH, F128r, F128i);
    // x_f_g = gglu(sigmoid(gate0)*|f2|) -> cat2 (slots 7-8) [0:128)
    se_gate_h(st, S(5), CCH, P(IN_FF_G0_W1), P(IN_FF_G0_B1), P(IN_FF_G0_BNG), P(IN_FF_G0_BNB),
              P(IN_FF_G0_W2), P(IN_FF_G0_B2), CCH, S(1), glin, gmid, bmu, brs);
    k_gate_abs_gglu_embed<<<gb(N1, 256), 256, 0, st>>>(S(1), S(5), S(6), S(7), CCH, 2 * CCH, 0, N1);
    // x_s_g = gglu(dw1(y2)) -> cat2 [128:256)
    k_grouped_conv<<<gb(N1, 256), 256, 0, st>>>(S(0), CCH, P(IN_FF_DW1), CCH, CCH, 3, 1, S(7), 2 * CCH, CCH);
    k_gglu_embed<<<gb(N1, 256), 256, 0, st>>>(S(7), CCH, 2 * CCH, CCH, N1);
    // x5_g, x7_g -> cat6 (slots 9..14, 768 channels) [512:640), [640:768)
    k_grouped_conv<<<gb(N1, 256), 256, 0, st>>>(S(0), CCH, P(IN_FF_DW5), CCH, CCH, 5, 2, S(9), 6 * CCH, 4 * CCH);
    k_gglu_embed<<<gb(N1, 256), 256, 0, st>>>(S(9), CCH, 6 * CCH, 4 * CCH, N1);
    k_grouped_conv<<<gb(N1, 256), 256, 0, st>>>(S(0), CCH, P(IN_FF_DW7), CCH, CCH, 7, 3, S(9), 6 * CCH, 5 * CCH);
    k_gglu_embed<<<gb(N1, 256), 256, 0, st>>>(S(9), CCH, 6 * CCH, 5 * CCH, N1);
    // xs2 = dw3(cat2) -> cat6 [256:512)
    k_grouped_conv<<<gb(N2, 256), 256, 0, st>>>(S(7), 2 * CCH, P(IN_FF_DW3), 2 * CCH, CCH, 3, 1, S(9), 6 * CCH, 2 * CCH);
    // f3 = fft2(cat2) (256 ch): rows s1-2 / s5-6, f3 -> s15-16 / s17-18
    fft2_real(st, S(7), S(1), S(5), S(15), S(17), 2 * CCH, F128r, F128i);
    // gate1, gated complex -> s5-6 / s19-20
    se_gate_h(st, S(15), 2 * CCH, P(IN_FF_G1_W1), P(IN_FF_G1_B1), P(IN_FF_G1_BNG), P(IN_FF_G1_BNB),
              P(IN_FF_G1_W2), P(IN_FF_G1_B2), 2 * CCH, S(1), glin, gmid, bmu, brs);
    k_sig_cmul<<<gb(N2, 256), 256, 0, st>>>(S(1), S(15), S(17), S(5), S(19), N2);
    // xf2 = abs(ifft2 spatial) -> cat6 [0:256)
    fft2_cplx(st, S(5), S(19), S(15), S(17), S(5), S(19), 2 * CCH, iF128r, iF128i);
    k_abs_embed<<<gb(N2, 256), 256, 0, st>>>(S(5), S(19), S(9), 2 * CCH, 6 * CCH, 0, N2);
    // ffn_out = proj(cat6, 768 -> 128) -> s1 ; out = x1 + ffn_out
    conv1x1(st, S(9), 6 * CCH, P(IN_FF_PROJ), nullptr, S(1), CCH, CCH, 0);
    k_add2<<<gb(N1, 256), 256, 0, st>>>(S(4), S(1), (float*)d_out, N1);
}
